// DFT_19490561590113
// MI455X (gfx1250) — 8-bit matrix operand order repaired and checked by reading, compiled, not run on hardware
//
#include <hip/hip_runtime.h>
#include <stdint.h>

typedef __attribute__((ext_vector_type(8))) int v8i;

#define N_FFT        512
#define M_TOTAL      65536
#define M_BLOCK      32
#define PLANE_BYTES  (512 * 512)
#define WS_PLANES_OFF 256

__global__ void init_ws(unsigned int* __restrict__ ws) {
    if (threadIdx.x < 3) ws[threadIdx.x] = 0u;
}

__global__ void absmax_kernel(const float* __restrict__ p, int n,
                              unsigned int* __restrict__ slot) {
    __shared__ unsigned int red[256];
    unsigned int m = 0u;
    for (int i = blockIdx.x * blockDim.x + threadIdx.x; i < n;
         i += gridDim.x * blockDim.x) {
        m = max(m, __float_as_uint(fabsf(p[i])));
    }
    red[threadIdx.x] = m;
    __syncthreads();
    for (int s = 128; s > 0; s >>= 1) {
        if ((int)threadIdx.x < s)
            red[threadIdx.x] = max(red[threadIdx.x], red[threadIdx.x + s]);
        __syncthreads();
    }
    if (threadIdx.x == 0) atomicMax(slot, red[0]);
}

__global__ void pack_weights(const float* __restrict__ wr,
                             const float* __restrict__ wi,
                             const unsigned int* __restrict__ scales,
                             int8_t* __restrict__ planes) {
    const int id = blockIdx.x * blockDim.x + threadIdx.x;
    const int b    = id & 31;
    const int lane = (id >> 5) & 31;
    const int kc   = (id >> 10) & 7;
    const int nt   = id >> 13;
    const int V = b >> 2;
    const int k = kc * 64 + ((b >> 3) << 4) + ((lane >> 4) << 3) + (b & 7);
    const int n = (nt << 4) + (lane & 15);

    const float swr = __uint_as_float(scales[1]) * (1.0f / 32767.0f);
    const float swi = __uint_as_float(scales[2]) * (1.0f / 32767.0f);
    const float irr = (swr > 0.f) ? 1.0f / swr : 0.f;
    const float iri = (swi > 0.f) ? 1.0f / swi : 0.f;

    int wqr = (int)rintf(wr[k * 512 + n] * irr);
    wqr = min(max(wqr, -32767), 32767);
    int wqi = (int)rintf(wi[k * 512 + n] * iri);
    wqi = min(max(wqi, -32767), 32767);

    planes[0 * PLANE_BYTES + id] = (int8_t)(wqr >> 8);
    planes[1 * PLANE_BYTES + id] = (int8_t)(wqr & 255);
    planes[2 * PLANE_BYTES + id] = (int8_t)(wqi >> 8);
    planes[3 * PLANE_BYTES + id] = (int8_t)(wqi & 255);
}

__global__ void __launch_bounds__(256)
dft_gemm(const float* __restrict__ x,
         const int8_t* __restrict__ planes,
         const unsigned int* __restrict__ scales,
         float* __restrict__ out) {
    __shared__ __align__(16) int8_t ldsHi[M_BLOCK * 512];
    __shared__ __align__(16) int8_t ldsLo[M_BLOCK * 512];

    const float sx  = __uint_as_float(scales[0]) * (1.0f / 32767.0f);
    const float swr = __uint_as_float(scales[1]) * (1.0f / 32767.0f);
    const float swi = __uint_as_float(scales[2]) * (1.0f / 32767.0f);
    const float invSx = (sx > 0.f) ? 1.0f / sx : 0.f;

    const int mBase = blockIdx.x * M_BLOCK;

    for (int i = threadIdx.x; i < M_BLOCK * 128; i += 256) {
        const int m  = i >> 7;
        const int kq = (i & 127) << 2;
        const float4 v = *reinterpret_cast<const float4*>(
            &x[(size_t)(mBase + m) * 512 + kq]);
        const float f[4] = {v.x, v.y, v.z, v.w};
        unsigned int hiPack = 0u, loPack = 0u;
#pragma unroll
        for (int j = 0; j < 4; ++j) {
            int q = (int)rintf(f[j] * invSx);
            q = min(max(q, -32768), 32767);
            hiPack |= ((unsigned int)((q >> 8) & 255)) << (8 * j);
            loPack |= ((unsigned int)(q & 255)) << (8 * j);
        }
        *reinterpret_cast<unsigned int*>(&ldsHi[m * 512 + kq]) = hiPack;
        *reinterpret_cast<unsigned int*>(&ldsLo[m * 512 + kq]) = loPack;
    }
    __syncthreads();

    const int wave = threadIdx.x >> 5;
    const int lane = threadIdx.x & 31;
    const int row  = lane & 15;
    const int koff = (lane >> 4) << 3;

    const int8_t* pHiR = planes + 0 * PLANE_BYTES;
    const int8_t* pLoR = planes + 1 * PLANE_BYTES;
    const int8_t* pHiI = planes + 2 * PLANE_BYTES;
    const int8_t* pLoI = planes + 3 * PLANE_BYTES;

    for (int nt = wave; nt < 32; nt += 8) {
        v8i aRA0 = {0}, aRM0 = {0}, aRD0 = {0};
        v8i aIA0 = {0}, aIM0 = {0}, aID0 = {0};
        v8i aRA1 = {0}, aRM1 = {0}, aRD1 = {0};
        v8i aIA1 = {0}, aIM1 = {0}, aID1 = {0};

        for (int kc = 0; kc < 8; ++kc) {
            const size_t bbase = ((size_t)(nt * 8 + kc) * 32 + lane) * 32;
            union { v8i v; int4 i4[2]; } bHiR, bLoR, bHiI, bLoI;
            bHiR.i4[0] = *reinterpret_cast<const int4*>(pHiR + bbase);
            bHiR.i4[1] = *reinterpret_cast<const int4*>(pHiR + bbase + 16);
            bLoR.i4[0] = *reinterpret_cast<const int4*>(pLoR + bbase);
            bLoR.i4[1] = *reinterpret_cast<const int4*>(pLoR + bbase + 16);
            bHiI.i4[0] = *reinterpret_cast<const int4*>(pHiI + bbase);
            bHiI.i4[1] = *reinterpret_cast<const int4*>(pHiI + bbase + 16);
            bLoI.i4[0] = *reinterpret_cast<const int4*>(pLoI + bbase);
            bLoI.i4[1] = *reinterpret_cast<const int4*>(pLoI + bbase + 16);

            union { v8i v; unsigned long long q[4]; } fHi0, fLo0, fHi1, fLo1;
            const int ab0 = row * 512 + kc * 64 + koff;
            const int ab1 = (row + 16) * 512 + kc * 64 + koff;
#pragma unroll
            for (int p = 0; p < 4; ++p) {
                fHi0.q[p] = *reinterpret_cast<const unsigned long long*>(&ldsHi[ab0 + 16 * p]);
                fLo0.q[p] = *reinterpret_cast<const unsigned long long*>(&ldsLo[ab0 + 16 * p]);
                fHi1.q[p] = *reinterpret_cast<const unsigned long long*>(&ldsHi[ab1 + 16 * p]);
                fLo1.q[p] = *reinterpret_cast<const unsigned long long*>(&ldsLo[ab1 + 16 * p]);
            }

            aRA0 = __builtin_amdgcn_wmma_i32_16x16x64_iu8(true,  fHi0.v, true,  bHiR.v, aRA0, false, false);
            aRA1 = __builtin_amdgcn_wmma_i32_16x16x64_iu8(true,  fHi1.v, true,  bHiR.v, aRA1, false, false);
            aIA0 = __builtin_amdgcn_wmma_i32_16x16x64_iu8(true,  fHi0.v, true,  bHiI.v, aIA0, false, false);
            aIA1 = __builtin_amdgcn_wmma_i32_16x16x64_iu8(true,  fHi1.v, true,  bHiI.v, aIA1, false, false);

            aRM0 = __builtin_amdgcn_wmma_i32_16x16x64_iu8(true,  fHi0.v, false, bLoR.v, aRM0, false, false);
            aRM1 = __builtin_amdgcn_wmma_i32_16x16x64_iu8(true,  fHi1.v, false, bLoR.v, aRM1, false, false);
            aIM0 = __builtin_amdgcn_wmma_i32_16x16x64_iu8(true,  fHi0.v, false, bLoI.v, aIM0, false, false);
            aIM1 = __builtin_amdgcn_wmma_i32_16x16x64_iu8(true,  fHi1.v, false, bLoI.v, aIM1, false, false);
            aRM0 = __builtin_amdgcn_wmma_i32_16x16x64_iu8(false, fLo0.v, true,  bHiR.v, aRM0, false, false);
            aRM1 = __builtin_amdgcn_wmma_i32_16x16x64_iu8(false, fLo1.v, true,  bHiR.v, aRM1, false, false);
            aIM0 = __builtin_amdgcn_wmma_i32_16x16x64_iu8(false, fLo0.v, true,  bHiI.v, aIM0, false, false);
            aIM1 = __builtin_amdgcn_wmma_i32_16x16x64_iu8(false, fLo1.v, true,  bHiI.v, aIM1, false, false);

            aRD0 = __builtin_amdgcn_wmma_i32_16x16x64_iu8(false, fLo0.v, false, bLoR.v, aRD0, false, false);
            aRD1 = __builtin_amdgcn_wmma_i32_16x16x64_iu8(false, fLo1.v, false, bLoR.v, aRD1, false, false);
            aID0 = __builtin_amdgcn_wmma_i32_16x16x64_iu8(false, fLo0.v, false, bLoI.v, aID0, false, false);
            aID1 = __builtin_amdgcn_wmma_i32_16x16x64_iu8(false, fLo1.v, false, bLoI.v, aID1, false, false);
        }

        const float scR = sx * swr, scI = sx * swi;
        const int colBase = nt * 16 + (lane & 15);
        const int mOff = (lane >> 4) << 3;
#pragma unroll
        for (int r = 0; r < 8; ++r) {
            const size_t orow0 = (size_t)(mBase + r + mOff) * 512 + colBase;
            const size_t orow1 = (size_t)(mBase + 16 + r + mOff) * 512 + colBase;
            const float vR0 = 65536.0f * (float)aRA0[r] + 256.0f * (float)aRM0[r] + (float)aRD0[r];
            const float vI0 = 65536.0f * (float)aIA0[r] + 256.0f * (float)aIM0[r] + (float)aID0[r];
            const float vR1 = 65536.0f * (float)aRA1[r] + 256.0f * (float)aRM1[r] + (float)aRD1[r];
            const float vI1 = 65536.0f * (float)aIA1[r] + 256.0f * (float)aIM1[r] + (float)aID1[r];
            out[orow0] = vR0 * scR;
            out[(size_t)M_TOTAL * 512 + orow0] = vI0 * scI;
            out[orow1] = vR1 * scR;
            out[(size_t)M_TOTAL * 512 + orow1] = vI1 * scI;
        }
    }
}

extern "C" void kernel_launch(void* const* d_in, const int* in_sizes, int n_in,
                              void* d_out, int out_size, void* d_ws, size_t ws_size,
                              hipStream_t stream) {
    const float* x  = (const float*)d_in[0];
    const float* wr = (const float*)d_in[1];
    const float* wi = (const float*)d_in[2];
    float* out = (float*)d_out;

    unsigned int* scales = (unsigned int*)d_ws;
    int8_t* planes = (int8_t*)d_ws + WS_PLANES_OFF;

    init_ws<<<1, 32, 0, stream>>>(scales);
    absmax_kernel<<<1024, 256, 0, stream>>>(x, M_TOTAL * N_FFT, &scales[0]);
    absmax_kernel<<<256,  256, 0, stream>>>(wr, N_FFT * N_FFT, &scales[1]);
    absmax_kernel<<<256,  256, 0, stream>>>(wi, N_FFT * N_FFT, &scales[2]);
    pack_weights<<<1024, 256, 0, stream>>>(wr, wi, scales, planes);
    dft_gemm<<<M_TOTAL / M_BLOCK, 256, 0, stream>>>(x, planes, scales, out);
}
